// CPEN_80951543595194
// MI455X (gfx1250) — matrix operand order repaired and checked by reading, not compiled, not run on hardware
//
#include <hip/hip_runtime.h>
#include <hip/hip_bf16.h>

typedef __attribute__((ext_vector_type(16))) _Float16 v16h;
typedef __attribute__((ext_vector_type(8)))  _Float16 v8h;
typedef __attribute__((ext_vector_type(8)))  float    v8f;

#define HW 16384
#define IMW 128
#define ECC_EPS 1e-6f

// ---------------------------------------------------------------------------
// conv1: 1 -> 32 channels, 3x3 SAME, ReLU, f32 in -> f16 channels-last out
// h1 layout: [img][pix][ci]  (each pixel's 32 channels contiguous, 64B)
// ---------------------------------------------------------------------------
__global__ void conv1_kernel(const float* __restrict__ warped,
                             const float* __restrict__ tmpl,
                             const float* __restrict__ w1,
                             const float* __restrict__ b1,
                             _Float16* __restrict__ h1) {
    int gid = blockIdx.x * blockDim.x + threadIdx.x;   // 8*HW threads
    int img = gid / HW;
    int pix = gid - img * HW;
    int y = pix >> 7, x = pix & 127;
    const float* src = (img < 4) ? (warped + (size_t)img * HW)
                                 : (tmpl + (size_t)(img - 4) * HW);
    float patch[9];
#pragma unroll
    for (int dy = 0; dy < 3; ++dy)
#pragma unroll
        for (int dx = 0; dx < 3; ++dx) {
            int yy = y + dy - 1, xx = x + dx - 1;
            patch[dy * 3 + dx] =
                (yy >= 0 && yy < IMW && xx >= 0 && xx < IMW) ? src[yy * IMW + xx] : 0.0f;
        }
    _Float16* dst = h1 + ((size_t)img * HW + pix) * 32;
#pragma unroll
    for (int cc = 0; cc < 4; ++cc) {
        v8h o;
#pragma unroll
        for (int j = 0; j < 8; ++j) {
            int co = cc * 8 + j;
            float s = b1[co];
#pragma unroll
            for (int k = 0; k < 9; ++k) s += w1[co * 9 + k] * patch[k];
            o[j] = (_Float16)fmaxf(s, 0.0f);
        }
        *(v8h*)(dst + cc * 8) = o;   // 16B vector store
    }
}

// ---------------------------------------------------------------------------
// weight pack: [co][ci][3][3] f32 -> per-lane A-fragment image f16:
//   dst[((tap*2 + cotile)*32 + lane)*16 + v]
// A-matrix (16-bit 16x32): lane m = co = cotile*16 + (lane&15);
//   kbase = (lane<16)?0:16; element v -> ci = kbase + v.
// Each lane's 16 halfs are then one contiguous 32B load.
// ---------------------------------------------------------------------------
__global__ void pack_weights(const float* __restrict__ w2,
                             const float* __restrict__ w3,
                             _Float16* __restrict__ w2A,
                             _Float16* __restrict__ w3A) {
    int gid = blockIdx.x * blockDim.x + threadIdx.x;   // 2*9216
    if (gid >= 2 * 9216) return;
    const float* src = (gid < 9216) ? w2 : w3;
    _Float16* dst = (gid < 9216) ? w2A : w3A;
    int idx = (gid < 9216) ? gid : gid - 9216;
    int v = idx & 15;
    int lane = (idx >> 4) & 31;
    int tt = idx >> 9;           // tap*2 + cotile
    int cotile = tt & 1;
    int tap = tt >> 1;
    int kbase = (lane >= 16) ? 16 : 0;
    int ci = kbase + v;
    int co = cotile * 16 + (lane & 15);
    dst[idx] = (_Float16)src[co * 288 + ci * 9 + tap];
}

// ---------------------------------------------------------------------------
// WMMA conv (transposed GEMM): D[co][pix] = sum_k W[co][k] * X[k][pix].
// A = packed weights (contiguous per lane), B = channels-last activations
// (contiguous per lane), D stores are contiguous channel runs per pixel.
// Block = 128 thr (4 waves), grid = 8 imgs * 128 rows; each wave owns 32
// pixels (two 16-wide N-tiles) x all 32 co (two 16-wide M-tiles).
// All lanes always active -> EXEC all-1s at every WMMA.
// ---------------------------------------------------------------------------
template <bool RELU, bool OUT_F16>
__global__ void conv_wmma_kernel(const _Float16* __restrict__ in,   // [img][pix][32] f16
                                 const _Float16* __restrict__ wA,   // packed A fragments
                                 const float* __restrict__ bias,
                                 _Float16* __restrict__ out16,      // [img][pix][32] f16
                                 float* __restrict__ out32) {       // [img][pix][32] f32
    int row = blockIdx.x & 127;
    int img = blockIdx.x >> 7;
    int tid = threadIdx.x;
    int wid = tid >> 5;
    int lane = tid & 31;
    bool hi = lane >= 16;
    int ln = lane & 15;
    int mofs = hi ? 8 : 0;       // C/D row offset for this half-wave
    int kstart = hi ? 16 : 0;    // B-matrix K window for this half-wave
    const _Float16* inimg = in + (size_t)img * HW * 32;

    __builtin_amdgcn_s_wait_tensorcnt(0);   // CDNA5 split-counter path (no-op here)

    v8f acc[2][2] = {};   // [pixel tile][co tile]
    for (int tap = 0; tap < 9; ++tap) {
        int dy = tap / 3 - 1;
        int dx = tap - (tap / 3) * 3 - 1;
        // A fragments: one 32B contiguous load per co-tile
        v16h a0 = *(const v16h*)(wA + ((size_t)(tap * 2 + 0) * 32 + lane) * 16);
        v16h a1 = *(const v16h*)(wA + ((size_t)(tap * 2 + 1) * 32 + lane) * 16);
        int yy = row + dy;
#pragma unroll
        for (int t = 0; t < 2; ++t) {
            int xx = wid * 32 + t * 16 + ln + dx;
            v16h b = {};
            if (yy >= 0 && yy < IMW && xx >= 0 && xx < IMW)
                b = *(const v16h*)(inimg + ((size_t)yy * IMW + xx) * 32 + kstart);
            acc[t][0] = __builtin_amdgcn_wmma_f32_16x16x32_f16(false, a0, false, b,
                                                               (short)0, acc[t][0], false, false);
            acc[t][1] = __builtin_amdgcn_wmma_f32_16x16x32_f16(false, a1, false, b,
                                                               (short)0, acc[t][1], false, false);
        }
    }
    // D layout: lane -> pixel = tile_x0 + (lane&15); VGPR r -> co = r + mofs
    // (+16 for second co tile). 8 consecutive channels per lane per tile.
#pragma unroll
    for (int t = 0; t < 2; ++t) {
        int px = wid * 32 + t * 16 + ln;
        size_t pixbase = ((size_t)img * HW + (size_t)row * IMW + px) * 32;
#pragma unroll
        for (int ct = 0; ct < 2; ++ct) {
            int cobase = ct * 16 + mofs;
            if (OUT_F16) {
                v8h o;
#pragma unroll
                for (int r = 0; r < 8; ++r) {
                    float v = acc[t][ct][r] + bias[cobase + r];
                    if (RELU) v = fmaxf(v, 0.0f);
                    o[r] = (_Float16)v;
                }
                *(v8h*)(out16 + pixbase + cobase) = o;      // 16B store
            } else {
                v8f o;
#pragma unroll
                for (int r = 0; r < 8; ++r) {
                    float v = acc[t][ct][r] + bias[cobase + r];
                    if (RELU) v = fmaxf(v, 0.0f);
                    o[r] = v;
                }
                *(v8f*)(out32 + pixbase + cobase) = o;      // 32B store
            }
        }
    }
}

// ---------------------------------------------------------------------------
// prepA: materialize per-problem img & centered template (tz). One WG / bc.
// f3 is channels-last: fw[b][ch][pix] = f3[(b*HW+pix)*32+ch],
//                      ft[b][ch][pix] = f3[((4+b)*HW+pix)*32+ch].
// ---------------------------------------------------------------------------
__global__ void prepA_kernel(const float* __restrict__ warped,
                             const float* __restrict__ tmpl,
                             const float* __restrict__ f3,
                             float* __restrict__ imgs,
                             float* __restrict__ tz) {
    int bc = blockIdx.x;
    int b = bc / 33;
    int c = bc - b * 33;
    const float* srcI; size_t stI;
    if (c == 0) { srcI = warped + (size_t)b * HW;                stI = 1;  }
    else        { srcI = f3 + (size_t)b * HW * 32 + (c - 1);     stI = 32; }
    const float* srcT; size_t stT;
    if (c < 32) { srcT = f3 + (size_t)(4 + b) * HW * 32 + c;     stT = 32; }
    else        { srcT = tmpl + (size_t)b * HW;                  stT = 1;  }
    float* dI = imgs + (size_t)bc * HW;
    float* dT = tz + (size_t)bc * HW;
    int tid = threadIdx.x, lane = tid & 31, wid = tid >> 5;
    __shared__ float sW[8];
    __shared__ float sMean;
    float s = 0.0f;
    for (int i = tid; i < HW; i += 256) {
        dI[i] = srcI[(size_t)i * stI];
        float tv = srcT[(size_t)i * stT];
        dT[i] = tv;
        s += tv;
    }
    for (int off = 16; off > 0; off >>= 1) s += __shfl_down(s, off, 32);
    if (lane == 0) sW[wid] = s;
    __syncthreads();
    if (tid == 0) {
        float tot = 0.0f;
        for (int w = 0; w < 8; ++w) tot += sW[w];
        sMean = tot / (float)HW;
    }
    __syncthreads();
    float mean = sMean;
    for (int i = tid; i < HW; i += 256) dT[i] -= mean;
}

// ---------------------------------------------------------------------------
// prepB: jnp.gradient — central diff interior, one-sided at edges.
// ---------------------------------------------------------------------------
__global__ void prepB_kernel(const float* __restrict__ imgs,
                             float* __restrict__ gx, float* __restrict__ gy) {
    int bc = blockIdx.x;
    const float* im = imgs + (size_t)bc * HW;
    float* gxo = gx + (size_t)bc * HW;
    float* gyo = gy + (size_t)bc * HW;
    for (int i = threadIdx.x; i < HW; i += 256) {
        int y = i >> 7, x = i & 127;
        float gyv = (y == 0)   ? im[IMW + x] - im[x]
                  : (y == 127) ? im[127 * IMW + x] - im[126 * IMW + x]
                               : 0.5f * (im[(y + 1) * IMW + x] - im[(y - 1) * IMW + x]);
        float gxv = (x == 0)   ? im[y * IMW + 1] - im[y * IMW]
                  : (x == 127) ? im[y * IMW + 127] - im[y * IMW + 126]
                               : 0.5f * (im[y * IMW + x + 1] - im[y * IMW + x - 1]);
        gyo[i] = gyv;
        gxo[i] = gxv;
    }
}

// ---------------------------------------------------------------------------
// ECC Gauss-Newton: one 256-thread WG per problem, 5 iterations.
// 42 scalars reduced per iter: H(21), G^T*iw(6), G^T*1(6), G^T*tz(6),
// sum iw, sum iw^2, tz.iw.  6x6 Gauss-Jordan on thread 0.
// ---------------------------------------------------------------------------
__global__ void ecc_kernel(const float* __restrict__ imgs,
                           const float* __restrict__ tz,
                           const float* __restrict__ gxA,
                           const float* __restrict__ gyA,
                           float* __restrict__ pOut) {
    int bc = blockIdx.x;
    const float* im  = imgs + (size_t)bc * HW;
    const float* tzp = tz  + (size_t)bc * HW;
    const float* gxp = gxA + (size_t)bc * HW;
    const float* gyp = gyA + (size_t)bc * HW;
    int tid = threadIdx.x, lane = tid & 31, wid = tid >> 5;

    __shared__ float sP[6];
    __shared__ float sRed[8][48];
    __shared__ float sTot[48];
    if (tid < 6) sP[tid] = (tid == 0 || tid == 4) ? 1.0f : 0.0f;
    __syncthreads();

    for (int iter = 0; iter < 5; ++iter) {
        float p0 = sP[0], p1 = sP[1], p2 = sP[2];
        float p3 = sP[3], p4 = sP[4], p5 = sP[5];
        float acc[42];
#pragma unroll
        for (int q = 0; q < 42; ++q) acc[q] = 0.0f;

        for (int i = tid; i < HW; i += 256) {
            int yi = i >> 7, xi = i & 127;
            float xf = (float)xi, yf = (float)yi;
            float xw = p0 * xf + p1 * yf + p2;
            float yw = p3 * xf + p4 * yf + p5;
            float xc = fminf(fmaxf(xw, 0.0f), 127.0f);
            float yc = fminf(fmaxf(yw, 0.0f), 127.0f);
            float x0f = floorf(xc), y0f = floorf(yc);
            float wx = xc - x0f, wy = yc - y0f;
            int x0i = (int)x0f, y0i = (int)y0f;
            int x1i = min(x0i + 1, 127), y1i = min(y0i + 1, 127);
            int i00 = y0i * IMW + x0i, i01 = y0i * IMW + x1i;
            int i10 = y1i * IMW + x0i, i11 = y1i * IMW + x1i;
            float w00 = (1.0f - wx) * (1.0f - wy), w01 = wx * (1.0f - wy);
            float w10 = (1.0f - wx) * wy,          w11 = wx * wy;
            float iw = im[i00] * w00 + im[i01] * w01 + im[i10] * w10 + im[i11] * w11;
            float gx = gxp[i00] * w00 + gxp[i01] * w01 + gxp[i10] * w10 + gxp[i11] * w11;
            float gy = gyp[i00] * w00 + gyp[i01] * w01 + gyp[i10] * w10 + gyp[i11] * w11;
            float tzv = tzp[i];
            float G[6] = { gx * xf, gx * yf, gx, gy * xf, gy * yf, gy };
            int q = 0;
#pragma unroll
            for (int a = 0; a < 6; ++a)
#pragma unroll
                for (int b2 = a; b2 < 6; ++b2) { acc[q] += G[a] * G[b2]; ++q; }
#pragma unroll
            for (int a = 0; a < 6; ++a) {
                acc[21 + a] += G[a] * iw;
                acc[27 + a] += G[a];
                acc[33 + a] += G[a] * tzv;
            }
            acc[39] += iw;
            acc[40] += iw * iw;
            acc[41] += tzv * iw;
        }

#pragma unroll
        for (int q = 0; q < 42; ++q) {
            float v = acc[q];
            for (int off = 16; off > 0; off >>= 1) v += __shfl_down(v, off, 32);
            if (lane == 0) sRed[wid][q] = v;
        }
        __syncthreads();
        if (tid < 42) {
            float t2 = 0.0f;
#pragma unroll
            for (int w = 0; w < 8; ++w) t2 += sRed[w][tid];
            sTot[tid] = t2;
        }
        __syncthreads();

        if (tid == 0) {
            const float N = (float)HW;
            float Hm[6][6], Hi[6][6];
            int q = 0;
            for (int a = 0; a < 6; ++a)
                for (int b2 = a; b2 < 6; ++b2) {
                    float v = sTot[q++];
                    Hm[a][b2] = v;
                    Hm[b2][a] = v;
                }
            for (int a = 0; a < 6; ++a) Hm[a][a] += ECC_EPS;
            for (int a = 0; a < 6; ++a)
                for (int b2 = 0; b2 < 6; ++b2) Hi[a][b2] = (a == b2) ? 1.0f : 0.0f;
            for (int col = 0; col < 6; ++col) {
                float inv = 1.0f / Hm[col][col];
                for (int j = 0; j < 6; ++j) { Hm[col][j] *= inv; Hi[col][j] *= inv; }
                for (int r = 0; r < 6; ++r) {
                    if (r == col) continue;
                    float f = Hm[r][col];
                    for (int j = 0; j < 6; ++j) {
                        Hm[r][j] -= f * Hm[col][j];
                        Hi[r][j] -= f * Hi[col][j];
                    }
                }
            }
            float mean = sTot[39] / N;
            float Gi[6], Gt[6], HiGi[6];
            for (int a = 0; a < 6; ++a) {
                Gi[a] = sTot[21 + a] - mean * sTot[27 + a];
                Gt[a] = sTot[33 + a];
            }
            for (int a = 0; a < 6; ++a) {
                float s = 0.0f;
                for (int j = 0; j < 6; ++j) s += Hi[a][j] * Gi[j];
                HiGi[a] = s;
            }
            float GiHiGi = 0.0f, GtHiGi = 0.0f;
            for (int a = 0; a < 6; ++a) {
                GiHiGi += Gi[a] * HiGi[a];
                GtHiGi += Gt[a] * HiGi[a];
            }
            float iwz2 = sTot[40] - N * mean * mean;
            float num = iwz2 - GiHiGi;
            float den = sTot[41] - GtHiGi;
            float lam = num / ((fabsf(den) < ECC_EPS) ? ECC_EPS : den);
            for (int a = 0; a < 6; ++a) {
                float s = 0.0f;
                for (int j = 0; j < 6; ++j) s += Hi[a][j] * (lam * Gt[j] - Gi[j]);
                sP[a] += s;
            }
        }
        __syncthreads();
    }
    if (tid < 6) pOut[bc * 6 + tid] = sP[tid];
}

// ---------------------------------------------------------------------------
// finalize: softmax(raw_w, axis=channels) and weighted combine -> [4,2,3].
// ---------------------------------------------------------------------------
__global__ void finalize_kernel(const float* __restrict__ raw_w,
                                const float* __restrict__ p,
                                float* __restrict__ out) {
    int t = threadIdx.x;
    if (t >= 24) return;
    int b = t / 6, l = t - b * 6;
    float mx = -1e30f;
    for (int c = 0; c < 33; ++c) mx = fmaxf(mx, raw_w[c * 6 + l]);
    float den = 0.0f;
    for (int c = 0; c < 33; ++c) den += expf(raw_w[c * 6 + l] - mx);
    float s = 0.0f;
    for (int c = 0; c < 33; ++c) {
        float ws = expf(raw_w[c * 6 + l] - mx) / den;
        s += p[(b * 33 + c) * 6 + l] * ws;
    }
    out[t] = s;
}

// ---------------------------------------------------------------------------
extern "C" void kernel_launch(void* const* d_in, const int* in_sizes, int n_in,
                              void* d_out, int out_size, void* d_ws, size_t ws_size,
                              hipStream_t stream) {
    (void)in_sizes; (void)n_in; (void)out_size; (void)ws_size;
    const float* warped = (const float*)d_in[0];
    const float* tmpl   = (const float*)d_in[1];
    const float* w1     = (const float*)d_in[2];
    const float* b1     = (const float*)d_in[3];
    const float* w2     = (const float*)d_in[4];
    const float* b2     = (const float*)d_in[5];
    const float* w3     = (const float*)d_in[6];
    const float* b3     = (const float*)d_in[7];
    const float* raw_w  = (const float*)d_in[8];
    float* out = (float*)d_out;

    char* ws = (char*)d_ws;
    size_t off = 0;
    _Float16* h1  = (_Float16*)(ws + off); off += (size_t)8 * HW * 32 * 2;  // 8 MiB, ch-last
    _Float16* h2  = (_Float16*)(ws + off); off += (size_t)8 * HW * 32 * 2;  // 8 MiB, ch-last
    float*    f3  = (float*)   (ws + off); off += (size_t)8 * HW * 32 * 4;  // 16 MiB, ch-last
    _Float16* w2A = (_Float16*)(ws + off); off += 32768;
    _Float16* w3A = (_Float16*)(ws + off); off += 32768;
    float*    imgs= (float*)   (ws + off); off += (size_t)132 * HW * 4;
    float*    tzb = (float*)   (ws + off); off += (size_t)132 * HW * 4;
    float*    gxb = (float*)   (ws + off); off += (size_t)132 * HW * 4;
    float*    gyb = (float*)   (ws + off); off += (size_t)132 * HW * 4;
    float*    pbf = (float*)   (ws + off); off += 132 * 6 * 4;

    // 1) pack conv2/conv3 weights into per-lane WMMA A fragments (f16)
    pack_weights<<<(2 * 9216 + 255) / 256, 256, 0, stream>>>(w2, w3, w2A, w3A);
    // 2) conv1 (8 images), channels-last f16 out
    conv1_kernel<<<(8 * HW) / 256, 256, 0, stream>>>(warped, tmpl, w1, b1, h1);
    // 3) conv2: WMMA, ReLU, f16 channels-last out
    conv_wmma_kernel<true, true><<<8 * 128, 128, 0, stream>>>(h1, w2A, b2, h2, nullptr);
    // 4) conv3: WMMA, no ReLU, f32 channels-last out
    conv_wmma_kernel<false, false><<<8 * 128, 128, 0, stream>>>(h2, w3A, b3, nullptr, f3);
    // 5) materialize per-problem image + centered template
    prepA_kernel<<<132, 256, 0, stream>>>(warped, tmpl, f3, imgs, tzb);
    // 6) gradients
    prepB_kernel<<<132, 256, 0, stream>>>(imgs, gxb, gyb);
    // 7) ECC alignment, 5 Gauss-Newton iterations per problem
    ecc_kernel<<<132, 256, 0, stream>>>(imgs, tzb, gxb, gyb, pbf);
    // 8) softmax combine -> [4,2,3]
    finalize_kernel<<<1, 32, 0, stream>>>(raw_w, pbf, out);
}
